// User_block_89197880803732
// MI455X (gfx1250) — matrix operand order repaired and checked by reading, not compiled, not run on hardware
//
#include <hip/hip_runtime.h>

// ---------------------------------------------------------------------------
// Types for CDNA5 WMMA
// ---------------------------------------------------------------------------
typedef __attribute__((ext_vector_type(16))) __bf16 v16bf;
typedef __attribute__((ext_vector_type(8)))  float  v8f;

#define BATCH 2048
#define NODES 64
#define IN_F  256
#define OUT_F 128
#define HID   512
#define COMBF (OUT_F * 3 + IN_F)   // 640
#define N_JOBS 64                  // 62 user nodes + irs + bs
#define USER_JOBS 62
#define EPS 1e-5f

// ---------------------------------------------------------------------------
// bf16 helpers (stored as unsigned short; RNE conversion)
// ---------------------------------------------------------------------------
__device__ __forceinline__ unsigned short f2bf(float f) {
    unsigned int u = __builtin_bit_cast(unsigned int, f);
    u = (u + 0x7FFFu + ((u >> 16) & 1u)) >> 16;
    return (unsigned short)u;
}
__device__ __forceinline__ float bf2f(unsigned short h) {
    unsigned int u = ((unsigned int)h) << 16;
    return __builtin_bit_cast(float, u);
}

// ---------------------------------------------------------------------------
// fp32 -> bf16 conversion (grid-stride)
// ---------------------------------------------------------------------------
__global__ void k_f32_to_bf16(const float* __restrict__ src,
                              unsigned short* __restrict__ dst,
                              long long n) {
    long long stride = (long long)gridDim.x * blockDim.x;
    for (long long i = (long long)blockIdx.x * blockDim.x + threadIdx.x; i < n; i += stride)
        dst[i] = f2bf(src[i]);
}

// ---------------------------------------------------------------------------
// Generic bf16 WMMA GEMM:  C[job, m, n] = sum_k A[m,k] * W[n,k] + bias[n]
//   - block: 128 threads = 4 waves; block tile 64(M) x 64(N); wave tile 32x32
//   - mode1: A is the bf16 copy of x [2048,64,256]; gather node column per job
//   - else : A = base + job*aJobStride, row stride = K
//   - weights/bias chosen per job: job < userJobs -> (Wu,bu); ==userJobs -> irs; else bs
// ---------------------------------------------------------------------------
__global__ void k_wmma_gemm(const unsigned short* __restrict__ A,
                            const unsigned short* __restrict__ Wu,
                            const unsigned short* __restrict__ Wi,
                            const unsigned short* __restrict__ Wb,
                            const float* __restrict__ bu,
                            const float* __restrict__ bi,
                            const float* __restrict__ bb,
                            unsigned short* __restrict__ C,
                            int M, int N, int K,
                            int userJobs, int mode1,
                            long long aJobStride, int aRowStride) {
    const int job = blockIdx.z;

    const unsigned short* W;
    const float* bias;
    if (job < userJobs)        { W = Wu; bias = bu; }
    else if (job == userJobs)  { W = Wi; bias = bi; }
    else                       { W = Wb; bias = bb; }

    const unsigned short* Abase;
    long long colOff;
    if (mode1) {
        // node index: j<3 -> j ; 3<=j<62 -> j+1 ; 62 -> 62 ; 63 -> 63
        int node = (job < 3) ? job : ((job < 62) ? job + 1 : job);
        Abase  = A;
        colOff = (long long)node * IN_F;
    } else {
        Abase  = A + (long long)job * aJobStride;
        colOff = 0;
    }

    const int lane = threadIdx.x & 31;
    const int wave = threadIdx.x >> 5;
    const int half = lane >> 4;       // 0 or 1
    const int l15  = lane & 15;

    const int tileM = blockIdx.y * 64 + (wave >> 1) * 32;
    const int tileN = blockIdx.x * 64 + (wave & 1) * 32;

    v8f acc[2][2] = {};

    for (int k0 = 0; k0 < K; k0 += 32) {
        union { unsigned int u[8]; v16bf v; } fa[2], fb[2];

        // ---- A fragments (16x32 bf16, layout per ISA 7.12.2) ----
        #pragma unroll
        for (int mi = 0; mi < 2; ++mi) {
            long long row = (long long)(tileM + mi * 16 + l15);
            const unsigned short* ap =
                Abase + row * (long long)aRowStride + colOff + k0 + half * 8;
            #pragma unroll
            for (int v = 0; v < 8; ++v) {
                int kk = (v < 4) ? 2 * v : 16 + 2 * (v - 4);
                fa[mi].u[v] = *(const unsigned int*)(ap + kk);  // bf16 pair
            }
        }
        // ---- B fragments (32x16 bf16 from row-major W[N,K]) ----
        #pragma unroll
        for (int ni = 0; ni < 2; ++ni) {
            long long nrow = (long long)(tileN + ni * 16 + l15);
            const unsigned short* bp = W + nrow * (long long)K + k0 + half * 8;
            #pragma unroll
            for (int v = 0; v < 8; ++v)
                fb[ni].u[v] = *(const unsigned int*)(bp + ((v < 4) ? 2 * v : 16 + 2 * (v - 4)));
        }
        // ---- 2x2 WMMAs ----
        #pragma unroll
        for (int mi = 0; mi < 2; ++mi)
            #pragma unroll
            for (int ni = 0; ni < 2; ++ni)
                acc[mi][ni] = __builtin_amdgcn_wmma_f32_16x16x32_bf16(
                    false, fa[mi].v, false, fb[ni].v,
                    (short)0, acc[mi][ni], false, false);
    }

    // ---- store (C/D layout: vgpr r -> M = r + 8*half, N = lane&15) ----
    #pragma unroll
    for (int mi = 0; mi < 2; ++mi) {
        #pragma unroll
        for (int ni = 0; ni < 2; ++ni) {
            int gn = tileN + ni * 16 + l15;
            float bv = bias[gn];
            #pragma unroll
            for (int r = 0; r < 8; ++r) {
                int gm = tileM + mi * 16 + r + half * 8;
                long long o = ((long long)job * M + gm) * (long long)N + gn;
                C[o] = f2bf(acc[mi][ni][r] + bv);
            }
        }
    }
}

// ---------------------------------------------------------------------------
// Per-column batch stats: one 256-thread block per (job, n) column over M rows
// stats[col*2] = mean, stats[col*2+1] = rsqrt(var + eps)   (biased var)
// ---------------------------------------------------------------------------
__global__ void k_col_stats(const unsigned short* __restrict__ X,
                            float* __restrict__ stats, int M, int N) {
    const int col = blockIdx.x;           // job*N + n
    const int n   = col % N;
    const long long base = (long long)(col / N) * M * N + n;

    float s = 0.f, sq = 0.f;
    for (int b = threadIdx.x; b < M; b += 256) {
        float v = bf2f(X[base + (long long)b * N]);
        s += v; sq += v * v;
    }
    __shared__ float sh_s[256];
    __shared__ float sh_q[256];
    sh_s[threadIdx.x] = s;
    sh_q[threadIdx.x] = sq;
    __syncthreads();
    for (int off = 128; off > 0; off >>= 1) {
        if (threadIdx.x < off) {
            sh_s[threadIdx.x] += sh_s[threadIdx.x + off];
            sh_q[threadIdx.x] += sh_q[threadIdx.x + off];
        }
        __syncthreads();
    }
    if (threadIdx.x == 0) {
        float mean = sh_s[0] / (float)M;
        float var  = sh_q[0] / (float)M - mean * mean;
        stats[col * 2 + 0] = mean;
        stats[col * 2 + 1] = rsqrtf(var + EPS);
    }
}

// ---------------------------------------------------------------------------
// Fused BatchNorm (affine) + sigmoid, elementwise (in-place OK).
// gamma/beta chosen per job class. outF32 -> write fp32, else bf16.
// ---------------------------------------------------------------------------
__global__ void k_bn_sigmoid(const unsigned short* __restrict__ X, void* Out,
                             int outF32, const float* __restrict__ stats,
                             const float* g0, const float* g1, const float* g2,
                             const float* e0, const float* e1, const float* e2,
                             int userJobs, int M, int N, long long total) {
    long long stride = (long long)gridDim.x * blockDim.x;
    for (long long i = (long long)blockIdx.x * blockDim.x + threadIdx.x;
         i < total; i += stride) {
        int n = (int)(i % N);
        long long row = i / N;
        int job = (int)(row / M);
        const float* g; const float* e;
        if (job < userJobs)       { g = g0; e = e0; }
        else if (job == userJobs) { g = g1; e = e1; }
        else                      { g = g2; e = e2; }
        int col = job * N + n;
        float mean = stats[col * 2 + 0];
        float rstd = stats[col * 2 + 1];
        float v = bf2f(X[i]);
        float y = (v - mean) * rstd * g[n] + e[n];
        y = 1.f / (1.f + __expf(-y));
        if (outF32) ((float*)Out)[i] = y;
        else        ((unsigned short*)Out)[i] = f2bf(y);
    }
}

// ---------------------------------------------------------------------------
// Build combined [2048, 640] bf16:
//  [0,128)   mean over jobs 0..61 of O2act
//  [128,256) O2act job 62 (irs) ; [256,384) O2act job 63 (bs)
//  [384,640) x[:, ORDER=3, :] (fp32 -> bf16)
// ---------------------------------------------------------------------------
__global__ void k_build_combined(const unsigned short* __restrict__ O2,
                                 const float* __restrict__ x,
                                 unsigned short* __restrict__ comb) {
    int i = blockIdx.x * blockDim.x + threadIdx.x;
    if (i >= BATCH * COMBF) return;
    int b = i / COMBF, c = i % COMBF;
    float v;
    if (c < OUT_F) {
        float s = 0.f;
        #pragma unroll 4
        for (int j = 0; j < USER_JOBS; ++j)
            s += bf2f(O2[((long long)j * BATCH + b) * OUT_F + c]);
        v = s * (1.0f / (float)USER_JOBS);
    } else if (c < 2 * OUT_F) {
        v = bf2f(O2[((long long)62 * BATCH + b) * OUT_F + (c - OUT_F)]);
    } else if (c < 3 * OUT_F) {
        v = bf2f(O2[((long long)63 * BATCH + b) * OUT_F + (c - 2 * OUT_F)]);
    } else {
        v = x[((long long)b * NODES + 3) * IN_F + (c - 3 * OUT_F)];
    }
    comb[i] = f2bf(v);
}

// ---------------------------------------------------------------------------
// Host orchestration
// ---------------------------------------------------------------------------
static inline size_t alignUp(size_t v) { return (v + 255) & ~(size_t)255; }

extern "C" void kernel_launch(void* const* d_in, const int* in_sizes, int n_in,
                              void* d_out, int out_size, void* d_ws, size_t ws_size,
                              hipStream_t stream) {
    (void)in_sizes; (void)n_in; (void)out_size; (void)ws_size;

    const float* x = (const float*)d_in[0];
    // param tuples: (W1,b1,g1,be1,W2,b2,g2,be2)
    // user: 1..8  irs: 9..16  bs: 17..24  comb: 25..32
    const float* uW1 = (const float*)d_in[1];  const float* ub1 = (const float*)d_in[2];
    const float* ug1 = (const float*)d_in[3];  const float* ue1 = (const float*)d_in[4];
    const float* uW2 = (const float*)d_in[5];  const float* ub2 = (const float*)d_in[6];
    const float* ug2 = (const float*)d_in[7];  const float* ue2 = (const float*)d_in[8];
    const float* iW1 = (const float*)d_in[9];  const float* ib1 = (const float*)d_in[10];
    const float* ig1 = (const float*)d_in[11]; const float* ie1 = (const float*)d_in[12];
    const float* iW2 = (const float*)d_in[13]; const float* ib2 = (const float*)d_in[14];
    const float* ig2 = (const float*)d_in[15]; const float* ie2 = (const float*)d_in[16];
    const float* bW1 = (const float*)d_in[17]; const float* bb1 = (const float*)d_in[18];
    const float* bg1 = (const float*)d_in[19]; const float* be1 = (const float*)d_in[20];
    const float* bW2 = (const float*)d_in[21]; const float* bb2 = (const float*)d_in[22];
    const float* bg2 = (const float*)d_in[23]; const float* be2 = (const float*)d_in[24];
    const float* cW1 = (const float*)d_in[25]; const float* cb1 = (const float*)d_in[26];
    const float* cg1 = (const float*)d_in[27]; const float* ce1 = (const float*)d_in[28];
    const float* cW2 = (const float*)d_in[29]; const float* cb2 = (const float*)d_in[30];
    const float* cg2 = (const float*)d_in[31]; const float* ce2 = (const float*)d_in[32];

    // ---- workspace carve-up (bytes) ----
    char* ws = (char*)d_ws;
    size_t off = 0;
    auto carve = [&](size_t bytes) { char* p = ws + off; off += alignUp(bytes); return p; };

    unsigned short* Xbf  = (unsigned short*)carve((size_t)BATCH * NODES * IN_F * 2);
    unsigned short* Wu1b = (unsigned short*)carve((size_t)HID * IN_F * 2);
    unsigned short* Wi1b = (unsigned short*)carve((size_t)HID * IN_F * 2);
    unsigned short* Wb1b = (unsigned short*)carve((size_t)HID * IN_F * 2);
    unsigned short* Wu2b = (unsigned short*)carve((size_t)OUT_F * HID * 2);
    unsigned short* Wi2b = (unsigned short*)carve((size_t)OUT_F * HID * 2);
    unsigned short* Wb2b = (unsigned short*)carve((size_t)OUT_F * HID * 2);
    unsigned short* Wc1b = (unsigned short*)carve((size_t)HID * COMBF * 2);
    unsigned short* Wc2b = (unsigned short*)carve((size_t)OUT_F * HID * 2);
    unsigned short* H1   = (unsigned short*)carve((size_t)N_JOBS * BATCH * HID * 2);
    unsigned short* O2   = (unsigned short*)carve((size_t)N_JOBS * BATCH * OUT_F * 2);
    unsigned short* COMB = (unsigned short*)carve((size_t)BATCH * COMBF * 2);
    unsigned short* H3   = (unsigned short*)carve((size_t)BATCH * HID * 2);
    unsigned short* O4   = (unsigned short*)carve((size_t)BATCH * OUT_F * 2);
    float* st1 = (float*)carve((size_t)N_JOBS * HID * 2 * 4);
    float* st2 = (float*)carve((size_t)N_JOBS * OUT_F * 2 * 4);
    float* st3 = (float*)carve((size_t)HID * 2 * 4);
    float* st4 = (float*)carve((size_t)OUT_F * 2 * 4);

    // ---- 1) convert x and weights to bf16 ----
    k_f32_to_bf16<<<4096, 256, 0, stream>>>(x, Xbf, (long long)BATCH * NODES * IN_F);
    k_f32_to_bf16<<<512, 256, 0, stream>>>(uW1, Wu1b, (long long)HID * IN_F);
    k_f32_to_bf16<<<512, 256, 0, stream>>>(iW1, Wi1b, (long long)HID * IN_F);
    k_f32_to_bf16<<<512, 256, 0, stream>>>(bW1, Wb1b, (long long)HID * IN_F);
    k_f32_to_bf16<<<256, 256, 0, stream>>>(uW2, Wu2b, (long long)OUT_F * HID);
    k_f32_to_bf16<<<256, 256, 0, stream>>>(iW2, Wi2b, (long long)OUT_F * HID);
    k_f32_to_bf16<<<256, 256, 0, stream>>>(bW2, Wb2b, (long long)OUT_F * HID);
    k_f32_to_bf16<<<1280, 256, 0, stream>>>(cW1, Wc1b, (long long)HID * COMBF);
    k_f32_to_bf16<<<256, 256, 0, stream>>>(cW2, Wc2b, (long long)OUT_F * HID);

    // ---- 2) stage 1: H1 = x_node @ W1^T + b1  (64 jobs) ----
    k_wmma_gemm<<<dim3(HID / 64, BATCH / 64, N_JOBS), 128, 0, stream>>>(
        Xbf, Wu1b, Wi1b, Wb1b, ub1, ib1, bb1, H1,
        BATCH, HID, IN_F, USER_JOBS, /*mode1=*/1, 0, NODES * IN_F);
    k_col_stats<<<N_JOBS * HID, 256, 0, stream>>>(H1, st1, BATCH, HID);
    k_bn_sigmoid<<<8192, 256, 0, stream>>>(
        H1, H1, 0, st1, ug1, ig1, bg1, ue1, ie1, be1,
        USER_JOBS, BATCH, HID, (long long)N_JOBS * BATCH * HID);

    // ---- 3) stage 2: O2 = H1act @ W2^T + b2 ----
    k_wmma_gemm<<<dim3(OUT_F / 64, BATCH / 64, N_JOBS), 128, 0, stream>>>(
        H1, Wu2b, Wi2b, Wb2b, ub2, ib2, bb2, O2,
        BATCH, OUT_F, HID, USER_JOBS, /*mode1=*/0, (long long)BATCH * HID, HID);
    k_col_stats<<<N_JOBS * OUT_F, 256, 0, stream>>>(O2, st2, BATCH, OUT_F);
    k_bn_sigmoid<<<4096, 256, 0, stream>>>(
        O2, O2, 0, st2, ug2, ig2, bg2, ue2, ie2, be2,
        USER_JOBS, BATCH, OUT_F, (long long)N_JOBS * BATCH * OUT_F);

    // ---- 4) combined = [mean(user), irs, bs, x[:,3,:]] ----
    k_build_combined<<<(BATCH * COMBF + 255) / 256, 256, 0, stream>>>(O2, x, COMB);

    // ---- 5) comb net layer 1 ----
    k_wmma_gemm<<<dim3(HID / 64, BATCH / 64, 1), 128, 0, stream>>>(
        COMB, Wc1b, Wc1b, Wc1b, cb1, cb1, cb1, H3,
        BATCH, HID, COMBF, /*userJobs=*/1, /*mode1=*/0, 0, COMBF);
    k_col_stats<<<HID, 256, 0, stream>>>(H3, st3, BATCH, HID);
    k_bn_sigmoid<<<2048, 256, 0, stream>>>(
        H3, H3, 0, st3, cg1, cg1, cg1, ce1, ce1, ce1,
        1, BATCH, HID, (long long)BATCH * HID);

    // ---- 6) comb net layer 2 -> final fp32 output ----
    k_wmma_gemm<<<dim3(OUT_F / 64, BATCH / 64, 1), 128, 0, stream>>>(
        H3, Wc2b, Wc2b, Wc2b, cb2, cb2, cb2, O4,
        BATCH, OUT_F, HID, /*userJobs=*/1, /*mode1=*/0, 0, HID);
    k_col_stats<<<OUT_F, 256, 0, stream>>>(O4, st4, BATCH, OUT_F);
    k_bn_sigmoid<<<1024, 256, 0, stream>>>(
        O4, d_out, 1, st4, cg2, cg2, cg2, ce2, ce2, ce2,
        1, BATCH, OUT_F, (long long)BATCH * OUT_F);
}
